// CascadeCodebookCluster_53644141527043
// MI455X (gfx1250) — matrix operand order repaired and checked by reading, not compiled, not run on hardware
//
#include <hip/hip_runtime.h>

typedef __attribute__((ext_vector_type(16))) __bf16 v16bf;
typedef __attribute__((ext_vector_type(8)))  float  v8f;
typedef __attribute__((ext_vector_type(4)))  float  v4f;

#define N_TOK 65536
#define EMB_D 256
#define M0C 2197
#define M1C 169
#define M2C 13
#define P0C 2208   /* padded to 16 */
#define P1C 176
#define P2C 16
#define PTOT (P0C + P1C + P2C)   /* 2400 padded codebook rows total */

__device__ __forceinline__ unsigned short f32_to_bf16_rne(float x) {
  unsigned u = __float_as_uint(x);
  unsigned r = u + 0x7fffu + ((u >> 16) & 1u);   // round-to-nearest-even
  return (unsigned short)(r >> 16);
}
__device__ __forceinline__ float bf16s_to_f32(unsigned short s) {
  return __uint_as_float(((unsigned)s) << 16);
}

// ---------------------------------------------------------------------------
// Prep: for each padded codebook row, write bf16 hi/lo planes + ||w||^2.
// One wave (32 lanes) per row; pad rows get zeros and wnorm = +inf.
// ---------------------------------------------------------------------------
__global__ __launch_bounds__(256) void vq_prep(
    const float* __restrict__ w0, const float* __restrict__ w1,
    const float* __restrict__ w2,
    unsigned short* __restrict__ whi, unsigned short* __restrict__ wlo,
    float* __restrict__ wnorm)
{
  int wave = (blockIdx.x * 256 + (int)threadIdx.x) >> 5;
  int lane = threadIdx.x & 31;
  if (wave >= PTOT) return;

  const float* src; int m, row;
  if (wave < P0C)            { src = w0; m = M0C; row = wave; }
  else if (wave < P0C + P1C) { src = w1; m = M1C; row = wave - P0C; }
  else                       { src = w2; m = M2C; row = wave - (P0C + P1C); }

  unsigned short* hrow = whi + (size_t)wave * EMB_D;
  unsigned short* lrow = wlo + (size_t)wave * EMB_D;

  if (row < m) {
    const float* rp = src + (size_t)row * EMB_D;
    float acc = 0.f;
#pragma unroll
    for (int t = 0; t < 2; ++t) {
      v4f x = *(const v4f*)(rp + t * 128 + lane * 4);
#pragma unroll
      for (int c = 0; c < 4; ++c) {
        float f = x[c];
        acc += f * f;
        unsigned short hb = f32_to_bf16_rne(f);
        float lo = f - bf16s_to_f32(hb);
        hrow[t * 128 + lane * 4 + c] = hb;
        lrow[t * 128 + lane * 4 + c] = f32_to_bf16_rne(lo);
      }
    }
#pragma unroll
    for (int off = 16; off > 0; off >>= 1) acc += __shfl_xor(acc, off, 32);
    if (lane == 0) wnorm[wave] = acc;
  } else {
#pragma unroll
    for (int t = 0; t < 2; ++t)
#pragma unroll
      for (int c = 0; c < 4; ++c) {
        hrow[t * 128 + lane * 4 + c] = 0;
        lrow[t * 128 + lane * 4 + c] = 0;
      }
    if (lane == 0) wnorm[wave] = __builtin_inff();
  }
}

// ---------------------------------------------------------------------------
// Main: each wave owns TWO 16-row embed tiles (32 rows); A hi/lo bf16 splits
// (K=256) for both tiles held in VGPRs across all column chunks of all 3
// layers, so every B fragment load feeds 6 WMMAs (2 tiles x bf16x3 terms).
// Distances via v_wmma_f32_16x16x32_bf16, per-lane running argmin, width-16
// butterfly reduce, then fp32 row gather to output.
// ---------------------------------------------------------------------------
__global__ __launch_bounds__(256) void vq_main(
    const float* __restrict__ embeds,
    const float* __restrict__ w0f, const float* __restrict__ w1f,
    const float* __restrict__ w2f,
    const unsigned short* __restrict__ whi,
    const unsigned short* __restrict__ wlo,
    const float* __restrict__ wnorm,
    float* __restrict__ out)
{
  const int lane = threadIdx.x & 31;
  const int wid  = threadIdx.x >> 5;
  const int rowbase = blockIdx.x * 256 + wid * 32;   // 8 waves * 32 rows
  const int r = lane & 15;
  const int h = lane >> 4;

  union AF { v16bf v; unsigned short s[16]; };

  // --- Load both A-tiles (2 x 16 rows x 256 K) once; split to bf16 hi/lo.
  // 16-bit A layout per 32-wide K-chunk: half h holds K = h*16 + {0..15}
  // in regs 0..7 (contiguous, matching B).
  v16bf a_hi[2][8], a_lo[2][8];
#pragma unroll
  for (int t = 0; t < 2; ++t) {
    const float* erow = embeds + (size_t)(rowbase + t * 16 + r) * EMB_D;
#pragma unroll
    for (int kc = 0; kc < 8; ++kc) {
      AF th, tl;
#pragma unroll
      for (int half = 0; half < 2; ++half) {
        const float* p = erow + kc * 32 + h * 16 + half * 8;
        v4f x0 = *(const v4f*)(p);
        v4f x1 = *(const v4f*)(p + 4);
#pragma unroll
        for (int jj = 0; jj < 4; ++jj) {
          float f = x0[jj];
          unsigned short hb = f32_to_bf16_rne(f);
          th.s[half * 8 + jj] = hb;
          tl.s[half * 8 + jj] = f32_to_bf16_rne(f - bf16s_to_f32(hb));
        }
#pragma unroll
        for (int jj = 0; jj < 4; ++jj) {
          float f = x1[jj];
          unsigned short hb = f32_to_bf16_rne(f);
          th.s[half * 8 + 4 + jj] = hb;
          tl.s[half * 8 + 4 + jj] = f32_to_bf16_rne(f - bf16s_to_f32(hb));
        }
      }
      a_hi[t][kc] = th.v;
      a_lo[t][kc] = tl.v;
    }
  }

  const int nch[3]  = {P0C / 16, P1C / 16, P2C / 16};
  const int roff[3] = {0, P0C, P0C + P1C};

#pragma unroll
  for (int l = 0; l < 3; ++l) {
    float dmin[2][8];
    int   imin[2][8];
#pragma unroll
    for (int t = 0; t < 2; ++t)
#pragma unroll
      for (int v = 0; v < 8; ++v) { dmin[t][v] = 3.4e38f; imin[t][v] = 0x7fffffff; }

    const unsigned short* bh_base = whi + (size_t)roff[l] * EMB_D;
    const unsigned short* bl_base = wlo + (size_t)roff[l] * EMB_D;
    const float* wn = wnorm + roff[l];
    const int chunks = nch[l];

    for (int j = 0; j < chunks; ++j) {
      const int col = j * 16 + r;          // B layout: lane holds column r
      const unsigned short* bhp = bh_base + (size_t)col * EMB_D + h * 16;
      const unsigned short* blp = bl_base + (size_t)col * EMB_D + h * 16;

      // Pull next chunk's B rows toward the WGP while this chunk computes.
      if (j + 1 < chunks) {
        __builtin_prefetch(bhp + (size_t)16 * EMB_D, 0, 3);
        __builtin_prefetch(blp + (size_t)16 * EMB_D, 0, 3);
      }

      v8f c0 = {0.f, 0.f, 0.f, 0.f, 0.f, 0.f, 0.f, 0.f};
      v8f c1 = {0.f, 0.f, 0.f, 0.f, 0.f, 0.f, 0.f, 0.f};
#pragma unroll
      for (int kc = 0; kc < 8; ++kc) {
        v16bf bh = *(const v16bf*)(bhp + kc * 32);
        v16bf bl = *(const v16bf*)(blp + kc * 32);
        // Each B load pair feeds 6 WMMAs: 2 row tiles x (hi*hi, hi*lo, lo*hi).
        c0 = __builtin_amdgcn_wmma_f32_16x16x32_bf16(false, a_hi[0][kc], false, bh,
                                                     (short)0, c0, false, false);
        c1 = __builtin_amdgcn_wmma_f32_16x16x32_bf16(false, a_hi[1][kc], false, bh,
                                                     (short)0, c1, false, false);
        c0 = __builtin_amdgcn_wmma_f32_16x16x32_bf16(false, a_hi[0][kc], false, bl,
                                                     (short)0, c0, false, false);
        c1 = __builtin_amdgcn_wmma_f32_16x16x32_bf16(false, a_hi[1][kc], false, bl,
                                                     (short)0, c1, false, false);
        c0 = __builtin_amdgcn_wmma_f32_16x16x32_bf16(false, a_lo[0][kc], false, bh,
                                                     (short)0, c0, false, false);
        c1 = __builtin_amdgcn_wmma_f32_16x16x32_bf16(false, a_lo[1][kc], false, bh,
                                                     (short)0, c1, false, false);
      }
      const float wnv = wn[col];
#pragma unroll
      for (int v = 0; v < 8; ++v) {
        float d0 = wnv - 2.0f * c0[v];     // ||w||^2 - 2 e.w (||e||^2 const)
        float d1 = wnv - 2.0f * c1[v];
        if (d0 < dmin[0][v]) { dmin[0][v] = d0; imin[0][v] = col; }
        if (d1 < dmin[1][v]) { dmin[1][v] = d1; imin[1][v] = col; }
      }
    }

    // Butterfly min+argmin across the 16 lanes holding columns 0..15
    // (index tie-break = smaller index, matching argmin's first-min rule).
#pragma unroll
    for (int t = 0; t < 2; ++t)
#pragma unroll
      for (int v = 0; v < 8; ++v) {
#pragma unroll
        for (int off = 1; off < 16; off <<= 1) {
          float od = __shfl_xor(dmin[t][v], off, 16);
          int   oi = __shfl_xor(imin[t][v], off, 16);
          if (od < dmin[t][v] || (od == dmin[t][v] && oi < imin[t][v])) {
            dmin[t][v] = od; imin[t][v] = oi;
          }
        }
      }

    // Gather winning fp32 codebook rows -> output (wave-cooperative float4).
    const float* wsrc = (l == 0) ? w0f : (l == 1) ? w1f : w2f;
    float* obase = out + ((size_t)l * N_TOK + rowbase) * EMB_D;
#pragma unroll
    for (int rr = 0; rr < 32; ++rr) {
      int t = rr >> 4;
      int rt = rr & 15;
      int v = rt & 7;
      int srcLane = (rt >> 3) * 16;        // C layout: row rt lives in half rt>>3
      int gid = __shfl(imin[t][v], srcLane, 32);
      const v4f* s = (const v4f*)(wsrc + (size_t)gid * EMB_D);
      v4f* dp = (v4f*)(obase + (size_t)rr * EMB_D);
      dp[lane]      = s[lane];
      dp[lane + 32] = s[lane + 32];
    }
  }
}

// ---------------------------------------------------------------------------
extern "C" void kernel_launch(void* const* d_in, const int* in_sizes, int n_in,
                              void* d_out, int out_size, void* d_ws, size_t ws_size,
                              hipStream_t stream)
{
  (void)in_sizes; (void)n_in; (void)out_size; (void)ws_size;

  const float* embeds = (const float*)d_in[0];
  const float* w0 = (const float*)d_in[1];
  const float* w1 = (const float*)d_in[2];
  const float* w2 = (const float*)d_in[3];

  char* ws = (char*)d_ws;
  const size_t planeBytes = (size_t)PTOT * EMB_D * sizeof(unsigned short); // 1,228,800
  unsigned short* whi = (unsigned short*)ws;
  unsigned short* wlo = (unsigned short*)(ws + planeBytes);
  float*          wnorm = (float*)(ws + 2 * planeBytes);

  vq_prep<<<PTOT / 8, 256, 0, stream>>>(w0, w1, w2, whi, wlo, wnorm);
  vq_main<<<N_TOK / 256, 256, 0, stream>>>(embeds, w0, w1, w2,
                                           whi, wlo, wnorm, (float*)d_out);
}
